// conv_mask_20426864459877
// MI455X (gfx1250) — matrix operand order repaired and checked by reading, not compiled, not run on hardware
//
#include <hip/hip_runtime.h>

typedef __attribute__((ext_vector_type(16))) __bf16 v16bf;
typedef __attribute__((ext_vector_type(16))) short  v16s;
typedef __attribute__((ext_vector_type(8)))  float  v8f;
typedef __attribute__((ext_vector_type(4)))  unsigned int v4u;
typedef __attribute__((ext_vector_type(8)))  int    v8i;
typedef __attribute__((ext_vector_type(4)))  int    v4i;

#define BS   32
#define CIN  512
#define COUT 512
#define H    28
#define W    28
#define HW   784
#define HP   30            /* padded spatial size      */
#define HWP  900           /* 30*30                    */
#define K9   4608          /* CIN*3*3                  */
#define NPOS (BS * HW)     /* 25088                    */
#define NSTEP 144          /* K9 / 32                  */

// ---------------------------------------------------------------------------
// fp32 -> bf16, round-to-nearest-even
// ---------------------------------------------------------------------------
__device__ __forceinline__ unsigned short f2bf(float f) {
  union { float f; unsigned u; } v; v.f = f;
  unsigned u = v.u;
  unsigned r = u + 0x7FFFu + ((u >> 16) & 1u);
  if ((u & 0x7F800000u) == 0x7F800000u) r = u;  // inf/nan passthrough
  return (unsigned short)(r >> 16);
}

// ---------------------------------------------------------------------------
// Kernel 1: per-(b,c) centroid stats. One wave32 per (b,c) image.
// ---------------------------------------------------------------------------
__global__ __launch_bounds__(32)
void cm_stats_kernel(const float* __restrict__ x,
                     float* __restrict__ mu_xg, float* __restrict__ mu_yg) {
  const int bc   = blockIdx.x;
  const int lane = threadIdx.x;
  const float* p = x + (size_t)bc * HW;

  float s = 0.f, sx = 0.f, sy = 0.f;
  for (int i = lane; i < HW; i += 32) {
    float v = p[i];
    int h = i / W, w = i - h * W;
    s  += v;
    sx += (float)(h + 1) * v;
    sy += (float)(w + 1) * v;
  }
  #pragma unroll
  for (int off = 16; off > 0; off >>= 1) {
    s  += __shfl_xor(s,  off, 32);
    sx += __shfl_xor(sx, off, 32);
    sy += __shfl_xor(sy, off, 32);
  }
  if (lane == 0) {
    s = fmaxf(s, 1e-9f);
    float mux = fmaxf(rintf(sx / s), 1.0f);
    float muy = fmaxf(rintf(sy / s), 1.0f);
    int ix = (int)(mux - 1.0f); if (ix > H - 1) ix = H - 1;
    int iy = (int)(muy - 1.0f); if (iy > H - 1) iy = H - 1;
    mu_xg[bc] = -1.0f + (2.0f / (H - 1)) * (float)ix;
    mu_yg[bc] = -1.0f + (2.0f / (H - 1)) * (float)iy;
  }
}

// ---------------------------------------------------------------------------
// Kernel 2: zero-fill the padded activation buffer (dword stores)
// ---------------------------------------------------------------------------
__global__ __launch_bounds__(256)
void cm_zero_kernel(unsigned int* __restrict__ p, int n32) {
  int i = blockIdx.x * 256 + threadIdx.x;
  int stride = gridDim.x * 256;
  for (; i < n32; i += stride) p[i] = 0u;
}

// ---------------------------------------------------------------------------
// Kernel 3: weight fp32 -> bf16, re-laid-out as [oc][kh*3+kw][ic]
// ---------------------------------------------------------------------------
__global__ __launch_bounds__(256)
void cm_wcvt_kernel(const float* __restrict__ w,
                    unsigned short* __restrict__ wbf, int n) {
  int i = blockIdx.x * 256 + threadIdx.x;
  if (i >= n) return;
  int oc = i / K9;
  int r  = i - oc * K9;
  int t  = r >> 9;            // kh*3+kw  (0..8)
  int ic = r & 511;
  wbf[i] = f2bf(w[(size_t)(oc * CIN + ic) * 9 + t]);
}

// ---------------------------------------------------------------------------
// Kernel 4: mask + ReLU + fp32 -> bf16 into zero-padded [b][c][30][30]
// ---------------------------------------------------------------------------
__global__ __launch_bounds__(256)
void cm_maskrelu_kernel(const float* __restrict__ x,
                        const float* __restrict__ mu_xg,
                        const float* __restrict__ mu_yg,
                        const float* __restrict__ mw,
                        const float* __restrict__ filt,
                        unsigned short* __restrict__ xrp) {
  int idx = blockIdx.x * 256 + threadIdx.x;
  if (idx >= BS * CIN * HW) return;
  int w  = idx % W;
  int h  = (idx / W) % H;
  int bc = idx / HW;
  int c  = bc % CIN;
  float posX = -1.0f + (2.0f / (H - 1)) * (float)h;
  float posY = -1.0f + (2.0f / (H - 1)) * (float)w;
  float m = 1.0f - (fabsf(posX - mu_xg[bc]) + fabsf(posY - mu_yg[bc])) * mw[c];
  m = fmaxf(m, -1.0f);
  if (filt[c] != 1.0f) m = 1.0f;
  float v = fmaxf(x[idx] * m, 0.0f);
  xrp[(size_t)bc * HWP + (h + 1) * HP + (w + 1)] = f2bf(v);
}

// ---------------------------------------------------------------------------
// Kernel 5: implicit-GEMM 3x3 conv via V_WMMA_F32_16X16X32_BF16, software-
// pipelined with double-buffered LDS:
//   - A tile (128x32 bf16) DMA'd by the Tensor Data Mover (pad_enable gives
//     the 72B LDS row stride); one DMA kept in flight (s_wait_tensorcnt 1).
//   - B tile im2col gather pipelined through registers: loads for step s+1
//     are issued before the step-s compute and stored to LDS next iteration.
// ---------------------------------------------------------------------------
__global__ __launch_bounds__(256)
void cm_conv_wmma_kernel(const unsigned short* __restrict__ wbf,
                         const unsigned short* __restrict__ xrp,
                         const float* __restrict__ bias,
                         float* __restrict__ out) {
  __shared__ unsigned short Asm[2][128 * 36];   // TDM-padded, 9216B each
  __shared__ unsigned short Bsm[2][32 * 64];    // fragment layout, 4096B each

  const int tid  = threadIdx.x;
  const int lane = tid & 31;
  const int wave = tid >> 5;                 // 0..7
  const int ocBlock = blockIdx.x * 128;
  const int nBlock  = blockIdx.y * 64;

  v8f acc0 = {}, acc1 = {}, acc2 = {}, acc3 = {};

  // ---- per-thread B-gather coordinates (step invariant) ----
  const int nloc = tid & 63;                 // owned n column
  const int k0   = tid >> 6;                 // 0..3 : k = k0 + 4*i
  size_t rowBase;
  {
    int ng  = nBlock + nloc;
    int b   = ng / HW;
    int pos = ng - b * HW;
    int oh  = pos / W;
    int ow  = pos - oh * W;
    rowBase = (size_t)(b * CIN) * HWP + oh * HP + ow;   // padded => no bounds
  }
  const int nPart = (((nloc >> 4) * 32 + (nloc & 15)) << 4);
  const unsigned short* bBase = xrp + rowBase + (size_t)k0 * HWP;

  // ---- static TDM descriptor pieces ----
  const unsigned long long aBytes =
      (unsigned long long)(const void*)(wbf + (size_t)ocBlock * K9);
  const unsigned ldsA = (unsigned)(unsigned long long)(uintptr_t)&Asm[0][0];
  const v8i g1 = { (1 << 16) | (1 << 20) | (3 << 22) | (1 << 25),
                   0, 16, 16 | (32 << 16), 128, 4608, 0x12000000, 0 };
  const v4i g2 = { 0, 0, 0, 0 };
  const v4i g3 = { 0, 0, 0, 0 };
  const v8i g4 = { 0, 0, 0, 0, 0, 0, 0, 0 };

  // B gather for a given K-step (t = step>>4 selects (kh,kw), 16 ic-steps each)
  auto bSrc = [&](int step) -> const unsigned short* {
    int t  = step >> 4;
    int kk = (step & 15) << 5;               // ic offset
    int kh = t / 3;
    int kw = t - kh * 3;
    return bBase + kh * HP + kw + (size_t)kk * HWP;
  };

  unsigned short breg[8];

  // ---- prologue: DMA A(0), gather B(0) into registers ----
  if (wave == 0) {
    v4u g0 = { 1u, ldsA, (unsigned)aBytes,
               (unsigned)(aBytes >> 32) | 0x80000000u /* type=2 */ };
    __builtin_amdgcn_tensor_load_to_lds(g0, g1, g2, g3, g4, 0);
  }
  {
    const unsigned short* src = bSrc(0);
    #pragma unroll
    for (int i = 0; i < 8; ++i) breg[i] = src[(size_t)i * 4 * HWP];
  }

  #pragma unroll 1
  for (int step = 0; step < NSTEP; ++step) {
    const int cur = step & 1;
    const int nxt = cur ^ 1;
    const bool more = (step + 1) < NSTEP;

    // ---- issue next A-tile DMA into the other buffer ----
    if (more && wave == 0) {
      unsigned long long ga = aBytes + (unsigned long long)(step + 1) * 64ull;
      v4u g0 = { 1u, ldsA + (unsigned)nxt * 9216u, (unsigned)ga,
                 (unsigned)(ga >> 32) | 0x80000000u };
      __builtin_amdgcn_tensor_load_to_lds(g0, g1, g2, g3, g4, 0);
    }

    // ---- publish current B tile (registers gathered last iteration) ----
    #pragma unroll
    for (int i = 0; i < 8; ++i) {
      int k = k0 + 4 * i;                    // 0..31
      Bsm[cur][nPart + (((k >> 3) & 1) << 8) + ((k >> 4) << 3) + (k & 7)] = breg[i];
    }

    // ---- issue next B gather (consumed after compute -> latency hidden) ----
    if (more) {
      const unsigned short* src = bSrc(step + 1);
      #pragma unroll
      for (int i = 0; i < 8; ++i) breg[i] = src[(size_t)i * 4 * HWP];
    }

    // ---- retire the older DMA: Asm[cur] complete (in-order TENSORcnt) ----
    if (wave == 0) {
      if (more) __builtin_amdgcn_s_wait_tensorcnt(1);
      else      __builtin_amdgcn_s_wait_tensorcnt(0);
    }
    __syncthreads();

    // ---- A fragment: ISA 16-bit A 16x32 layout ----
    v16s av;
    {
      const int row = wave * 16 + (lane & 15);
      const int kh8 = (lane >> 4) << 3;      // 0 or 8
      const uint2* ap2 = (const uint2*)&Asm[cur][row * 36 + kh8];
      union { v16s v; uint2 q[4]; } au;
      au.q[0] = ap2[0];                      // K = kh8+0..3
      au.q[1] = ap2[1];                      // K = kh8+4..7
      au.q[2] = ap2[4];                      // K = kh8+16..19
      au.q[3] = ap2[5];                      // K = kh8+20..23
      av = au.v;
    }

    // ---- 4 WMMAs: one A fragment vs 4 N-subtile B fragments ----
    #pragma unroll
    for (int j = 0; j < 4; ++j) {
      union { v16s v; uint4 q[2]; } bu;
      const uint4* bp = (const uint4*)&Bsm[cur][(j * 32 + lane) * 16];
      bu.q[0] = bp[0];
      bu.q[1] = bp[1];
      v8f* accp = (j == 0) ? &acc0 : (j == 1) ? &acc1
                : (j == 2) ? &acc2 : &acc3;
      *accp = __builtin_amdgcn_wmma_f32_16x16x32_bf16(
          false, __builtin_bit_cast(v16bf, av),
          false, __builtin_bit_cast(v16bf, bu.v),
          (short)0, *accp, false, false);
    }
    __syncthreads();
  }

  // ---- epilogue: C layout lane<16 -> M=v, lane>=16 -> M=v+8 ----
  const int ncol = lane & 15;
  const int mAdd = (lane >> 4) << 3;
  #pragma unroll
  for (int j = 0; j < 4; ++j) {
    const v8f acc = (j == 0) ? acc0 : (j == 1) ? acc1 : (j == 2) ? acc2 : acc3;
    int ng  = nBlock + j * 16 + ncol;
    int b   = ng / HW;
    int pos = ng - b * HW;
    #pragma unroll
    for (int v = 0; v < 8; ++v) {
      int oc = ocBlock + wave * 16 + mAdd + v;
      out[(size_t)((b * COUT + oc) * HW) + pos] = acc[v] + bias[oc];
    }
  }
}

// ---------------------------------------------------------------------------
// Launch
// ---------------------------------------------------------------------------
extern "C" void kernel_launch(void* const* d_in, const int* in_sizes, int n_in,
                              void* d_out, int out_size, void* d_ws, size_t ws_size,
                              hipStream_t stream) {
  const float* x           = (const float*)d_in[0];
  const float* weight      = (const float*)d_in[1];
  const float* bias        = (const float*)d_in[2];
  const float* mask_weight = (const float*)d_in[3];
  const float* filt        = (const float*)d_in[4];
  // d_in[5] = padding, fixed to 1 per the reference setup

  char* ws = (char*)d_ws;
  float* mu_xg = (float*)ws;
  float* mu_yg = mu_xg + BS * CIN;
  unsigned short* wbf = (unsigned short*)(ws + (size_t)2 * BS * CIN * sizeof(float));
  unsigned short* xrp = wbf + (size_t)COUT * K9;
  float* out = (float*)d_out;

  cm_stats_kernel<<<BS * CIN, 32, 0, stream>>>(x, mu_xg, mu_yg);

  int n32 = BS * CIN * HWP / 2;
  cm_zero_kernel<<<2048, 256, 0, stream>>>((unsigned int*)xrp, n32);

  int nw = COUT * K9;
  cm_wcvt_kernel<<<(nw + 255) / 256, 256, 0, stream>>>(weight, wbf, nw);

  int nx = BS * CIN * HW;
  cm_maskrelu_kernel<<<(nx + 255) / 256, 256, 0, stream>>>(
      x, mu_xg, mu_yg, mask_weight, filt, xrp);

  dim3 grid(COUT / 128, NPOS / 64);   // 4 x 392
  cm_conv_wmma_kernel<<<grid, 256, 0, stream>>>(wbf, xrp, bias, out);
}
